// acLSTM_50818053046809
// MI455X (gfx1250) — matrix operand order repaired and checked by reading, not compiled, not run on hardware
//
#include <hip/hip_runtime.h>
#include <hip/hip_bf16.h>

// ---------------------------------------------------------------------------
// acLSTM on MI455X (gfx1250): 3x LSTM(1024) + decoder(171), batch 64, T=100.
// bf16 WMMA (v_wmma_f32_16x16x32_bf16) with f32 accumulate; weights converted
// to bf16 once per launch (43MB -> resident in 192MB L2 for the recurrence).
// ---------------------------------------------------------------------------

typedef __attribute__((ext_vector_type(16))) __bf16 v16bf;
typedef __attribute__((ext_vector_type(8)))  float  v8f;

#define BATCH   64
#define HID     1024
#define FH      4096      // 4*HID
#define INSZ    171
#define KX1     192       // IN_SIZE padded to multiple of 32
#define TSTEPS  100
#define OUTSTR  (TSTEPS * INSZ)   // 17100, per-batch-row stride of d_out

__device__ __forceinline__ float sigmoidf_(float x) {
    return 1.0f / (1.0f + __expf(-x));
}

// ---- WMMA fragment loaders (CDNA5 ISA 7.12.2 layouts) ---------------------
// A: 16x32 bf16, row-major source, ld = row stride (elements).
// lane<16 : row=lane,    K = {k0..k0+7}   and {k0+16..k0+23}
// lane>=16: row=lane-16, K = {k0+8..+15}  and {k0+24..+31}
__device__ __forceinline__ v16bf load_frag_a(const __hip_bfloat16* __restrict__ X,
                                             int ld, int row0, int k0, int lane) {
    int row = row0 + (lane & 15);
    int kb  = k0 + ((lane >> 4) << 3);
    const __hip_bfloat16* p = X + (size_t)row * ld + kb;
    union { v16bf v; uint4 q[2]; } u;
    u.q[0] = *reinterpret_cast<const uint4*>(p);        // 8 bf16
    u.q[1] = *reinterpret_cast<const uint4*>(p + 16);   // 8 bf16
    return u.v;
}

// B: 32x16 bf16. W stored row-major [N_total, K] so B-column n == W row n.
// lane<16 : col=lane,    K = {k0..k0+7} and {k0+16..k0+23} ; lane>=16: K = {k0+8..+15} and {k0+24..+31}
__device__ __forceinline__ v16bf load_frag_b(const __hip_bfloat16* __restrict__ W,
                                             int ld, int col0, int k0, int lane) {
    int col = col0 + (lane & 15);
    int kb  = k0 + ((lane >> 4) << 3);
    const __hip_bfloat16* p = W + (size_t)col * ld + kb;
    union { v16bf v; uint4 q[2]; } u;
    u.q[0] = *reinterpret_cast<const uint4*>(p);
    u.q[1] = *reinterpret_cast<const uint4*>(p + 16);
    return u.v;
}

__device__ __forceinline__ v8f wmma_bf16(v16bf a, v16bf b, v8f c) {
    return __builtin_amdgcn_wmma_f32_16x16x32_bf16(
        /*neg_a=*/false, a, /*neg_b=*/false, b,
        /*c_mod=*/(short)0, c, /*reuse_a=*/false, /*reuse_b=*/false);
}

// ---------------------------------------------------------------------------
// Fused LSTM layer: gates = X@Wx^T + Hin@Wh^T + bsum ; cell update ; Hout(bf16)
// Grid: 32 blocks (32 hidden units each), 256 threads = 8 waves.
// Wave w: batch tile mt=w&3 (rows 16*mt), hidden subtile cs=w>>2 (16 cols).
// Each wave accumulates 4 gate tiles (i,f,g,o) over the SAME (batch,hidden)
// 16x16 tile -> the nonlinear cell update is purely per-lane on the C/D layout.
// ---------------------------------------------------------------------------
__global__ __launch_bounds__(256)
void lstm_layer_wmma(const __hip_bfloat16* __restrict__ X,  int kx,    // [64, kx]
                     const __hip_bfloat16* __restrict__ Wx,            // [4096, kx]
                     const __hip_bfloat16* __restrict__ Hin,           // [64, 1024]
                     const __hip_bfloat16* __restrict__ Wh,            // [4096, 1024]
                     const float* __restrict__ bsum,                   // [4096] b_ih+b_hh
                     float* __restrict__ C,                            // [64, 1024] in/out
                     __hip_bfloat16* __restrict__ Hout)                // [64, 1024]
{
    const int lane = threadIdx.x & 31;
    const int w    = threadIdx.x >> 5;
    const int mt   = w & 3;
    const int cs   = w >> 2;
    const int r0   = mt * 16;
    const int n0   = blockIdx.x * 32 + cs * 16;

    v8f acc[4];
    acc[0] = (v8f){}; acc[1] = (v8f){}; acc[2] = (v8f){}; acc[3] = (v8f){};

    // Phase 1: X @ Wx^T   (kx = 192 for layer 1, 1024 for layers 2/3)
    for (int k0 = 0; k0 < kx; k0 += 32) {
        v16bf a = load_frag_a(X, kx, r0, k0, lane);
#pragma unroll
        for (int g = 0; g < 4; ++g) {
            v16bf b = load_frag_b(Wx, kx, g * HID + n0, k0, lane);
            acc[g] = wmma_bf16(a, b, acc[g]);
        }
    }
    // Phase 2: Hin @ Wh^T  (K = 1024)
    for (int k0 = 0; k0 < HID; k0 += 32) {
        v16bf a = load_frag_a(Hin, HID, r0, k0, lane);
#pragma unroll
        for (int g = 0; g < 4; ++g) {
            v16bf b = load_frag_b(Wh, HID, g * HID + n0, k0, lane);
            acc[g] = wmma_bf16(a, b, acc[g]);
        }
    }

    // Cell update. C/D layout: vgpr r, lane l -> M = r + 8*(l>=16), N = l&15.
    const int n    = n0 + (lane & 15);
    const int boff = (lane >> 4) * 8;
    const float bi = bsum[n];
    const float bf = bsum[HID + n];
    const float bg = bsum[2 * HID + n];
    const float bo = bsum[3 * HID + n];
#pragma unroll
    for (int r = 0; r < 8; ++r) {
        const int brow = r0 + boff + r;
        const size_t idx = (size_t)brow * HID + n;
        float ig = sigmoidf_(acc[0][r] + bi);
        float fg = sigmoidf_(acc[1][r] + bf);
        float gg = tanhf   (acc[2][r] + bg);
        float og = sigmoidf_(acc[3][r] + bo);
        float c  = fg * C[idx] + ig * gg;
        C[idx]   = c;
        Hout[idx] = __float2bfloat16(og * tanhf(c));
    }
}

// ---------------------------------------------------------------------------
// Decoder: out = H2 @ Wd^T + bd  (N=171 padded to 176). Also prepares the
// next step's layer-1 input: x_next = cond(t+1) ? real_seq[:,t+1,:] : out.
// Grid: 11 blocks (16 cols each), 128 threads = 4 waves (one batch tile each).
// ---------------------------------------------------------------------------
__global__ __launch_bounds__(128)
void decoder_wmma(const __hip_bfloat16* __restrict__ H2,   // [64, 1024]
                  const __hip_bfloat16* __restrict__ Wd,   // [176, 1024] (rows>=171 zero)
                  const float* __restrict__ bd,            // [171]
                  float* __restrict__ Out,                 // [64, 17100]
                  const float* __restrict__ seq,           // [64, 100, 171]
                  __hip_bfloat16* __restrict__ Xnext,      // [64, 192]
                  int t)
{
    const int lane = threadIdx.x & 31;
    const int mt   = threadIdx.x >> 5;
    const int r0   = mt * 16;
    const int n0   = blockIdx.x * 16;

    v8f acc = (v8f){};
    for (int k0 = 0; k0 < HID; k0 += 32) {
        v16bf a = load_frag_a(H2, HID, r0, k0, lane);
        v16bf b = load_frag_b(Wd, HID, n0, k0, lane);
        acc = wmma_bf16(a, b, acc);
    }

    const int n = n0 + (lane & 15);
    if (n < INSZ) {
        const float bias = bd[n];
        const bool next_is_gt = (((t + 1) % 10) < 5);
        const int boff = (lane >> 4) * 8;
#pragma unroll
        for (int r = 0; r < 8; ++r) {
            const int brow = r0 + boff + r;
            const float o = acc[r] + bias;
            Out[(size_t)brow * OUTSTR + t * INSZ + n] = o;
            if (t + 1 < TSTEPS) {
                float xv = next_is_gt ? seq[((size_t)brow * TSTEPS + (t + 1)) * INSZ + n]
                                      : o;
                Xnext[(size_t)brow * KX1 + n] = __float2bfloat16(xv);
            }
        }
    }
}

// ---------------------------------------------------------------------------
// Helpers: weight f32->bf16 conversion with row/col zero padding, bias sums,
// state init (workspace is poisoned 0xAA by the harness before timing).
// ---------------------------------------------------------------------------
__global__ void convert_pad(const float* __restrict__ src, __hip_bfloat16* __restrict__ dst,
                            int rows_in, int kin, int kpad, int total) {
    int i = blockIdx.x * blockDim.x + threadIdx.x;
    if (i >= total) return;
    int r = i / kpad, k = i - r * kpad;
    float v = (r < rows_in && k < kin) ? src[(size_t)r * kin + k] : 0.0f;
    dst[i] = __float2bfloat16(v);
}

__global__ void bias_sum(const float* __restrict__ a, const float* __restrict__ b,
                         float* __restrict__ out, int n) {
    int i = blockIdx.x * blockDim.x + threadIdx.x;
    if (i < n) out[i] = a[i] + b[i];
}

__global__ void init_state(float* c0, float* c1, float* c2,
                           __hip_bfloat16* h0a, __hip_bfloat16* h0b,
                           __hip_bfloat16* h1a, __hip_bfloat16* h1b,
                           __hip_bfloat16* h2a, __hip_bfloat16* h2b,
                           __hip_bfloat16* x1, const float* __restrict__ seq) {
    int i = blockIdx.x * blockDim.x + threadIdx.x;
    if (i < BATCH * HID) {
        c0[i] = 0.0f; c1[i] = 0.0f; c2[i] = 0.0f;
        __hip_bfloat16 z = __float2bfloat16(0.0f);
        h0a[i] = z; h0b[i] = z; h1a[i] = z; h1b[i] = z; h2a[i] = z; h2b[i] = z;
    }
    if (i < BATCH * KX1) {       // cond[0] == 1 -> frame 0 from real_seq
        int b = i / KX1, k = i - b * KX1;
        x1[i] = __float2bfloat16(k < INSZ ? seq[(size_t)b * OUTSTR + k] : 0.0f);
    }
}

// ---------------------------------------------------------------------------
extern "C" void kernel_launch(void* const* d_in, const int* in_sizes, int n_in,
                              void* d_out, int out_size, void* d_ws, size_t ws_size,
                              hipStream_t stream) {
    const float* seq   = (const float*)d_in[0];
    const float* w_ih1 = (const float*)d_in[1];
    const float* w_hh1 = (const float*)d_in[2];
    const float* b_ih1 = (const float*)d_in[3];
    const float* b_hh1 = (const float*)d_in[4];
    const float* w_ih2 = (const float*)d_in[5];
    const float* w_hh2 = (const float*)d_in[6];
    const float* b_ih2 = (const float*)d_in[7];
    const float* b_hh2 = (const float*)d_in[8];
    const float* w_ih3 = (const float*)d_in[9];
    const float* w_hh3 = (const float*)d_in[10];
    const float* b_ih3 = (const float*)d_in[11];
    const float* b_hh3 = (const float*)d_in[12];
    const float* dec_w = (const float*)d_in[13];
    const float* dec_b = (const float*)d_in[14];
    float* out = (float*)d_out;

    // Carve workspace (256B-aligned slices).
    char* ws = (char*)d_ws;
    size_t off = 0;
    auto carve = [&](size_t bytes) -> char* {
        char* p = ws + off;
        off = (off + bytes + 255) & ~(size_t)255;
        return p;
    };
    __hip_bfloat16* wih1b = (__hip_bfloat16*)carve((size_t)FH * KX1 * 2);
    __hip_bfloat16* whh1b = (__hip_bfloat16*)carve((size_t)FH * HID * 2);
    __hip_bfloat16* wih2b = (__hip_bfloat16*)carve((size_t)FH * HID * 2);
    __hip_bfloat16* whh2b = (__hip_bfloat16*)carve((size_t)FH * HID * 2);
    __hip_bfloat16* wih3b = (__hip_bfloat16*)carve((size_t)FH * HID * 2);
    __hip_bfloat16* whh3b = (__hip_bfloat16*)carve((size_t)FH * HID * 2);
    __hip_bfloat16* decwb = (__hip_bfloat16*)carve((size_t)176 * HID * 2);
    float* bsum1 = (float*)carve(FH * 4);
    float* bsum2 = (float*)carve(FH * 4);
    float* bsum3 = (float*)carve(FH * 4);
    __hip_bfloat16* x1 = (__hip_bfloat16*)carve((size_t)BATCH * KX1 * 2);
    __hip_bfloat16* h0a = (__hip_bfloat16*)carve((size_t)BATCH * HID * 2);
    __hip_bfloat16* h0b = (__hip_bfloat16*)carve((size_t)BATCH * HID * 2);
    __hip_bfloat16* h1a = (__hip_bfloat16*)carve((size_t)BATCH * HID * 2);
    __hip_bfloat16* h1b = (__hip_bfloat16*)carve((size_t)BATCH * HID * 2);
    __hip_bfloat16* h2a = (__hip_bfloat16*)carve((size_t)BATCH * HID * 2);
    __hip_bfloat16* h2b = (__hip_bfloat16*)carve((size_t)BATCH * HID * 2);
    float* c0 = (float*)carve((size_t)BATCH * HID * 4);
    float* c1 = (float*)carve((size_t)BATCH * HID * 4);
    float* c2 = (float*)carve((size_t)BATCH * HID * 4);

    // 1) Convert weights to bf16 (padded); combine biases.
    {
        int tot = FH * KX1;
        convert_pad<<<(tot + 255) / 256, 256, 0, stream>>>(w_ih1, wih1b, FH, INSZ, KX1, tot);
        tot = FH * HID;
        convert_pad<<<(tot + 255) / 256, 256, 0, stream>>>(w_hh1, whh1b, FH, HID, HID, tot);
        convert_pad<<<(tot + 255) / 256, 256, 0, stream>>>(w_ih2, wih2b, FH, HID, HID, tot);
        convert_pad<<<(tot + 255) / 256, 256, 0, stream>>>(w_hh2, whh2b, FH, HID, HID, tot);
        convert_pad<<<(tot + 255) / 256, 256, 0, stream>>>(w_ih3, wih3b, FH, HID, HID, tot);
        convert_pad<<<(tot + 255) / 256, 256, 0, stream>>>(w_hh3, whh3b, FH, HID, HID, tot);
        tot = 176 * HID;
        convert_pad<<<(tot + 255) / 256, 256, 0, stream>>>(dec_w, decwb, INSZ, HID, HID, tot);
        bias_sum<<<FH / 256, 256, 0, stream>>>(b_ih1, b_hh1, bsum1, FH);
        bias_sum<<<FH / 256, 256, 0, stream>>>(b_ih2, b_hh2, bsum2, FH);
        bias_sum<<<FH / 256, 256, 0, stream>>>(b_ih3, b_hh3, bsum3, FH);
    }

    // 2) Zero h/c state, seed x1 with frame 0.
    init_state<<<(BATCH * HID + 255) / 256, 256, 0, stream>>>(
        c0, c1, c2, h0a, h0b, h1a, h1b, h2a, h2b, x1, seq);

    // 3) Recurrence: 4 stream-ordered kernels per timestep (graph-captured).
    for (int t = 0; t < TSTEPS; ++t) {
        const bool odd = (t & 1);
        const __hip_bfloat16* h0_in = odd ? h0b : h0a;
        __hip_bfloat16*       h0_out = odd ? h0a : h0b;
        const __hip_bfloat16* h1_in = odd ? h1b : h1a;
        __hip_bfloat16*       h1_out = odd ? h1a : h1b;
        const __hip_bfloat16* h2_in = odd ? h2b : h2a;
        __hip_bfloat16*       h2_out = odd ? h2a : h2b;

        lstm_layer_wmma<<<32, 256, 0, stream>>>(x1, KX1, wih1b, h0_in, whh1b,
                                                bsum1, c0, h0_out);
        lstm_layer_wmma<<<32, 256, 0, stream>>>(h0_out, HID, wih2b, h1_in, whh2b,
                                                bsum2, c1, h1_out);
        lstm_layer_wmma<<<32, 256, 0, stream>>>(h1_out, HID, wih3b, h2_in, whh3b,
                                                bsum3, c2, h2_out);
        decoder_wmma<<<11, 128, 0, stream>>>(h2_out, decwb, dec_b, out, seq, x1, t);
    }
}
